// QuantConv2d_49770081026769
// MI455X (gfx1250) — 8-bit matrix operand order repaired and checked by reading, compiled, not run on hardware
//
#include <hip/hip_runtime.h>
#include <stdint.h>

typedef int v8i __attribute__((ext_vector_type(8)));

#define EPSQ 1e-8f
#define NIMG 16
#define CIN  256
#define COUT 256
#define HH   64
#define WW   64
#define KTOT 2304
#define NKCHUNK 36

#define QW_OFF 4096
#define QX_OFF (4096 + COUT * KTOT)
#define QX_BYTES ((size_t)NIMG * HH * WW * CIN)

__global__ __launch_bounds__(256) void init_ws_kernel(int* __restrict__ I) {
  for (int i = threadIdx.x; i < 1024; i += 256) I[i] = 0;
}

__global__ __launch_bounds__(256) void maxabs_x_kernel(const float* __restrict__ x,
                                                       int n, int* __restrict__ I) {
  __shared__ float red[256];
  int t = threadIdx.x;
  float m = 0.f;
  for (int i = blockIdx.x * 256 + t; i < n; i += gridDim.x * 256)
    m = fmaxf(m, fabsf(x[i]));
  red[t] = m; __syncthreads();
  for (int s = 128; s > 0; s >>= 1) {
    if (t < s) red[t] = fmaxf(red[t], red[t + s]);
    __syncthreads();
  }
  if (t == 0) atomicMax(I, __float_as_int(red[0]));
}

__global__ __launch_bounds__(256) void wscale_kernel(const float* __restrict__ w,
                                                     float* __restrict__ F) {
  __shared__ float red[256];
  int o = blockIdx.x, t = threadIdx.x;
  float m = 0.f;
  for (int i = t; i < KTOT; i += 256)
    m = fmaxf(m, fabsf(w[(size_t)o * KTOT + i]));
  red[t] = m; __syncthreads();
  for (int s = 128; s > 0; s >>= 1) {
    if (t < s) red[t] = fmaxf(red[t], red[t + s]);
    __syncthreads();
  }
  if (t == 0) F[272 + o] = fmaxf(red[0], EPSQ) / 127.f;
}

__global__ __launch_bounds__(256) void finalize_kernel(const float* __restrict__ bias,
                                                       float* __restrict__ F,
                                                       const int* __restrict__ I) {
  int t = threadIdx.x;
  if (t == 0) {
    float xs = fmaxf(__int_as_float(I[0]), EPSQ) / 127.f;
    F[1] = xs;
    F[2] = 1.f / xs;
  }
  float bv = bias[t];
  float bs = fmaxf(fabsf(bv), EPSQ) / 127.f;
  float q  = fminf(fmaxf(rintf(bv / bs), -128.f), 127.f);
  F[528 + t] = q * bs;
}

__global__ __launch_bounds__(256) void quantw_kernel(const float* __restrict__ w,
                                                     const float* __restrict__ F,
                                                     signed char* __restrict__ qw) {
  int i = blockIdx.x * 256 + threadIdx.x;
  if (i >= COUT * KTOT) return;
  int o   = i / KTOT;
  int rem = i - o * KTOT;
  int c   = rem / 9;
  int r   = rem - c * 9;
  float sc = F[272 + o];
  float q  = fminf(fmaxf(rintf(w[i] / sc), -128.f), 127.f);
  qw[(size_t)o * KTOT + r * 256 + c] = (signed char)(int)q;
}

__global__ __launch_bounds__(256) void quantx_kernel(const float* __restrict__ x,
                                                     const float* __restrict__ F,
                                                     signed char* __restrict__ qx) {
  const float inv_xs = F[2];
  int pos = blockIdx.x * 256 + threadIdx.x;
  const float* px  = x + ((size_t)(pos >> 12) * CIN * 4096) + (size_t)(pos & 4095);
  uint32_t* qrow = (uint32_t*)(qx + (size_t)pos * CIN);
#pragma unroll 4
  for (int c0 = 0; c0 < CIN; c0 += 4) {
    uint32_t pk = 0;
#pragma unroll
    for (int j = 0; j < 4; ++j) {
      float xv = px[(size_t)(c0 + j) * 4096];
      float qf = fminf(fmaxf(rintf(xv * inv_xs), -128.f), 127.f);
      pk |= ((uint32_t)((int)qf & 0xFF)) << (8 * j);
    }
    qrow[c0 >> 2] = pk;
  }
}

template <bool PREQ>
__global__ __launch_bounds__(256) void conv_wmma_kernel(const float* __restrict__ x,
                                                        const signed char* __restrict__ qx,
                                                        const signed char* __restrict__ qw,
                                                        const float* __restrict__ F,
                                                        float* __restrict__ out) {
  __shared__ alignas(16) uint32_t As[2][64 * 16];
  __shared__ alignas(16) uint32_t Bs[2][128 * 16];

  const int tid  = threadIdx.x;
  const int lane = tid & 31;
  const int wv   = tid >> 5;
  const int nl   = lane & 15;
  const int half = lane >> 4;

  int bx = blockIdx.x;
  const int ntile = bx & 1;  bx >>= 1;
  const int h     = bx & 63; bx >>= 6;
  const int n     = bx;
  const int ocb   = ntile * 128;

  const float x_scale = F[1];
  const float inv_xs  = F[2];

  const int am = tid >> 2;
  const int aq = tid & 3;
  const int brow = tid >> 1;
  const int bh   = tid & 1;

  const uint32_t*    gw  = (const uint32_t*)(qw + (size_t)(ocb + brow) * KTOT) + bh * 8;
  const signed char* qxn = PREQ ? qx + (size_t)n * HH * WW * CIN + aq * 16 : nullptr;

  v8i acc[4];
#pragma unroll
  for (int mg = 0; mg < 4; ++mg) acc[mg] = (v8i){0, 0, 0, 0, 0, 0, 0, 0};

  uint4 ra, rb0, rb1;

  auto gload = [&](int ki) {
    const uint32_t* gsrc = gw + ki * 16;
    rb0 = *(const uint4*)(gsrc);
    rb1 = *(const uint4*)(gsrc + 4);
    __builtin_prefetch(gsrc + 16, 0, 1);
    const int r  = ki >> 2;
    const int c0 = (ki & 3) * 64;
    const int kh = r / 3;
    const int kw = r - kh * 3;
    const int ih = h + kh - 1;
    const int iw = am + kw - 1;
    const bool valid = ((unsigned)ih < 64u) & ((unsigned)iw < 64u);
    const int ihc = ih < 0 ? 0 : (ih > 63 ? 63 : ih);
    const int iwc = iw < 0 ? 0 : (iw > 63 ? 63 : iw);
    if (PREQ) {
      ra = *(const uint4*)(qxn + ((size_t)(ihc * WW + iwc)) * CIN + c0);
      if (!valid) ra = make_uint4(0u, 0u, 0u, 0u);
    } else {
      uint32_t d[4] = {0u, 0u, 0u, 0u};
      if (valid) {
#pragma unroll
        for (int dd = 0; dd < 4; ++dd) {
          uint32_t pk = 0;
#pragma unroll
          for (int j = 0; j < 4; ++j) {
            int c = c0 + aq * 16 + dd * 4 + j;
            float xv = x[(((size_t)n * CIN + c) * HH + ih) * WW + iw];
            float qf = fminf(fmaxf(rintf(xv * inv_xs), -128.f), 127.f);
            pk |= ((uint32_t)((int)qf & 0xFF)) << (8 * j);
          }
          d[dd] = pk;
        }
      }
      ra = make_uint4(d[0], d[1], d[2], d[3]);
    }
  };

  auto lds_store = [&](int buf) {
    *(uint4*)&Bs[buf][brow * 16 + bh * 8]     = rb0;
    *(uint4*)&Bs[buf][brow * 16 + bh * 8 + 4] = rb1;
    *(uint4*)&As[buf][am * 16 + aq * 4]       = ra;
  };

  auto compute = [&](int buf) {
    const uint32_t* bp = &Bs[buf][(wv * 16 + nl) * 16 + half * 4];
    uint4 b0 = *(const uint4*)(bp);
    uint4 b1 = *(const uint4*)(bp + 8);
    v8i bfrag;
    bfrag[0] = (int)b0.x; bfrag[1] = (int)b0.y; bfrag[2] = (int)b0.z; bfrag[3] = (int)b0.w;
    bfrag[4] = (int)b1.x; bfrag[5] = (int)b1.y; bfrag[6] = (int)b1.z; bfrag[7] = (int)b1.w;
#pragma unroll
    for (int mg = 0; mg < 4; ++mg) {
      const uint32_t* ap = &As[buf][(mg * 16 + nl) * 16 + half * 4];
      uint2 a0 = *(const uint2*)(ap);
      uint2 a1 = *(const uint2*)(ap + 2);
      uint2 a2 = *(const uint2*)(ap + 8);
      uint2 a3 = *(const uint2*)(ap + 10);
      v8i afrag;
      afrag[0] = (int)a0.x; afrag[1] = (int)a0.y;
      afrag[2] = (int)a1.x; afrag[3] = (int)a1.y;
      afrag[4] = (int)a2.x; afrag[5] = (int)a2.y;
      afrag[6] = (int)a3.x; afrag[7] = (int)a3.y;
      acc[mg] = __builtin_amdgcn_wmma_i32_16x16x64_iu8(true, afrag, true, bfrag,
                                                       acc[mg], false, false);
    }
  };

  gload(0);
#pragma unroll 2
  for (int ki = 0; ki < NKCHUNK; ++ki) {
    lds_store(ki & 1);
    __syncthreads();
    if (ki + 1 < NKCHUNK) gload(ki + 1);
    compute(ki & 1);
  }

  const int   oc  = ocb + wv * 16 + nl;
  const float wsc = F[272 + oc];
  const float bd  = F[528 + oc];
  const float sc  = x_scale * wsc;
  float* orow = out + (((size_t)n * COUT + oc) * HH + h) * WW;
#pragma unroll
  for (int mg = 0; mg < 4; ++mg) {
    float vals[8];
#pragma unroll
    for (int v = 0; v < 8; ++v) {
      float y = (float)acc[mg][v] * sc + bd;
      float q = fminf(fmaxf(rintf(y / wsc), -128.f), 127.f);
      vals[v] = q * wsc;
    }
    const int mbase = mg * 16 + half * 8;
    *(float4*)(orow + mbase)     = make_float4(vals[0], vals[1], vals[2], vals[3]);
    *(float4*)(orow + mbase + 4) = make_float4(vals[4], vals[5], vals[6], vals[7]);
  }
}

extern "C" void kernel_launch(void* const* d_in, const int* in_sizes, int n_in,
                              void* d_out, int out_size, void* d_ws, size_t ws_size,
                              hipStream_t stream) {
  (void)n_in; (void)out_size;
  const float* x    = (const float*)d_in[0];
  const float* w    = (const float*)d_in[1];
  const float* bias = (const float*)d_in[2];
  float*       out  = (float*)d_out;

  int*         I  = (int*)d_ws;
  float*       F  = (float*)d_ws;
  signed char* qw = (signed char*)d_ws + QW_OFF;
  signed char* qx = (signed char*)d_ws + QX_OFF;

  const int nx = in_sizes[0];
  const bool preq = ws_size >= (size_t)QX_OFF + QX_BYTES;

  init_ws_kernel<<<1, 256, 0, stream>>>(I);
  maxabs_x_kernel<<<1024, 256, 0, stream>>>(x, nx, I);
  wscale_kernel<<<COUT, 256, 0, stream>>>(w, F);
  finalize_kernel<<<1, 256, 0, stream>>>(bias, F, I);
  quantw_kernel<<<(COUT * KTOT + 255) / 256, 256, 0, stream>>>(w, F, qw);

  const int conv_grid = NIMG * HH * 2;
  if (preq) {
    quantx_kernel<<<(NIMG * HH * WW) / 256, 256, 0, stream>>>(x, F, qx);
    conv_wmma_kernel<true><<<conv_grid, 256, 0, stream>>>(x, qx, qw, F, out);
  } else {
    conv_wmma_kernel<false><<<conv_grid, 256, 0, stream>>>(x, qx, qw, F, out);
  }
}
